// MultiHeadAttention_37529424232674
// MI455X (gfx1250) — matrix operand order repaired and checked by reading, not compiled, not run on hardware
//
#include <hip/hip_runtime.h>
#include <hip/hip_bf16.h>

// ---------- types ----------
typedef __attribute__((ext_vector_type(16))) __bf16 v16bf;
typedef __attribute__((ext_vector_type(8)))  float  v8f;
typedef __attribute__((ext_vector_type(4)))  unsigned int u32x4;
typedef __attribute__((ext_vector_type(2)))  unsigned int u32x2;
typedef __attribute__((ext_vector_type(4)))  float  f32x4;

#define D_MODEL 1024
#define NHEAD   16
#define DHEAD   64
#define BATCH   8
#define SEQ     1024
#define NEGV    (-10000.0f)
#define LDSK    40   // padded LDS row stride (halves) for 32-wide K tiles

union H8 { u32x4 u; __bf16 h[8]; };
union H4 { u32x2 u; __bf16 h[4]; };

// ---------- CDNA5 async global->LDS copy (ASYNCcnt path), guarded ----------
#if defined(__gfx1250__) && __has_builtin(__builtin_amdgcn_global_load_async_to_lds_b128)
#define HAVE_ASYNC_LDS 1
// Builtin signature (from clang diagnostic): param0 = AS1 "int4*", param1 = AS3 "int4*".
typedef int v4i_vec __attribute__((vector_size(16)));
typedef v4i_vec __attribute__((address_space(1))) as1_v4i;
typedef v4i_vec __attribute__((address_space(3))) as3_v4i;
#define ASYNC_B128(gp, lp)                                                    \
  __builtin_amdgcn_global_load_async_to_lds_b128(                             \
      (as1_v4i*)(unsigned long long)(gp),                                     \
      (as3_v4i*)(unsigned int)(unsigned long long)(lp), 0, 0)
#if __has_builtin(__builtin_amdgcn_s_wait_asynccnt)
#define WAIT_ASYNC() __builtin_amdgcn_s_wait_asynccnt(0)
#else
#define WAIT_ASYNC() asm volatile("s_wait_asynccnt 0x0" ::: "memory")
#endif
#else
#define HAVE_ASYNC_LDS 0
#endif

// Build a v16bf WMMA fragment from two 16-byte LDS/global chunks.
static __device__ __forceinline__ v16bf frag_ld(const __bf16* lo, const __bf16* hi) {
  H8 a, b;
  a.u = *(const u32x4*)lo;
  b.u = *(const u32x4*)hi;
  v16bf f;
#pragma unroll
  for (int i = 0; i < 8; ++i) { f[i] = a.h[i]; f[i + 8] = b.h[i]; }
  return f;
}

static __device__ __forceinline__ v8f wmma_bf16(v16bf a, v16bf b, v8f c) {
  // (neg_a, A, neg_b, B, c_mod, C, reuse_a, reuse_b)
  return __builtin_amdgcn_wmma_f32_16x16x32_bf16(false, a, false, b, (short)0, c, false, false);
}

// =====================================================================
// Kernel 1: fused QKV projection.  C[M=8192,N=1024] = X @ W^T + b,
// written head-split [B,H,L,64] in bf16.  grid.z selects q/k/v.
// Block: 128 threads = 4 waves; block tile 128x128; wave tile 64x64.
// Register-prefetch double buffering: next tile's global loads issue
// before the WMMA section so HBM latency hides under the matrix ops.
// =====================================================================
__global__ __launch_bounds__(128)
void qkv_proj_kernel(const float* __restrict__ q, const float* __restrict__ k,
                     const float* __restrict__ v,
                     const float* __restrict__ wq, const float* __restrict__ bq,
                     const float* __restrict__ wk, const float* __restrict__ bk,
                     const float* __restrict__ wv, const float* __restrict__ bv,
                     __bf16* __restrict__ qh, __bf16* __restrict__ kh,
                     __bf16* __restrict__ vh)
{
  __shared__ __align__(16) __bf16 As[128][LDSK];
  __shared__ __align__(16) __bf16 Bs[128][LDSK];

  const int z = blockIdx.z;
  const float* X    = (z == 0) ? q  : (z == 1) ? k  : v;
  const float* W    = (z == 0) ? wq : (z == 1) ? wk : wv;
  const float* Bvec = (z == 0) ? bq : (z == 1) ? bk : bv;
  __bf16*      Out  = (z == 0) ? qh : (z == 1) ? kh : vh;

  const int m0   = blockIdx.x * 128;
  const int n0   = blockIdx.y * 128;
  const int t    = threadIdx.x;
  const int wave = t >> 5;
  const int lane = t & 31;
  const int l16  = lane & 15;
  const int g    = lane >> 4;
  const int wm   = (wave >> 1) * 64;
  const int wn   = (wave & 1) * 64;

  v8f acc[4][4];
#pragma unroll
  for (int mi = 0; mi < 4; ++mi)
#pragma unroll
    for (int ni = 0; ni < 4; ++ni) acc[mi][ni] = {};

  f32x4 pa[8], pb[8];
  auto tile_load = [&](int k0) {
#pragma unroll
    for (int i = 0; i < 8; ++i) {
      const int elem = (t + i * 128) * 4;
      const int row  = elem >> 5;
      const int col  = elem & 31;
      pa[i] = *(const f32x4*)&X[(size_t)(m0 + row) * D_MODEL + k0 + col];
      pb[i] = *(const f32x4*)&W[(size_t)(n0 + row) * D_MODEL + k0 + col];
    }
  };
  tile_load(0);

  for (int k0 = 0; k0 < D_MODEL; k0 += 32) {
    __syncthreads();
    // Drain prefetch registers into LDS (f32 -> bf16).
#pragma unroll
    for (int i = 0; i < 8; ++i) {
      const int elem = (t + i * 128) * 4;
      const int row  = elem >> 5;
      const int col  = elem & 31;
      H4 ha, hb;
#pragma unroll
      for (int j = 0; j < 4; ++j) { ha.h[j] = (__bf16)pa[i][j]; hb.h[j] = (__bf16)pb[i][j]; }
      *(u32x2*)&As[row][col] = ha.u;
      *(u32x2*)&Bs[row][col] = hb.u;
    }
    __syncthreads();
    if (k0 + 32 < D_MODEL) tile_load(k0 + 32);  // overlaps with WMMAs below

    v16bf af[4], bfr[4];
#pragma unroll
    for (int mi = 0; mi < 4; ++mi) {
      const __bf16* rowp = &As[wm + mi * 16 + l16][0];
      af[mi] = frag_ld(rowp + g * 8, rowp + 16 + g * 8);            // A layout
    }
#pragma unroll
    for (int ni = 0; ni < 4; ++ni) {
      const __bf16* rowp = &Bs[wn + ni * 16 + l16][0];
      bfr[ni] = frag_ld(rowp + g * 8, rowp + 16 + g * 8);           // B layout
    }
#pragma unroll
    for (int mi = 0; mi < 4; ++mi)
#pragma unroll
      for (int ni = 0; ni < 4; ++ni)
        acc[mi][ni] = wmma_bf16(af[mi], bfr[ni], acc[mi][ni]);
  }

  // Bias + head-split bf16 store.
#pragma unroll
  for (int mi = 0; mi < 4; ++mi) {
#pragma unroll
    for (int ni = 0; ni < 4; ++ni) {
      const int n  = n0 + wn + ni * 16 + l16;
      const float bias = Bvec[n];
      const int h = n >> 6, dd = n & 63;
#pragma unroll
      for (int r = 0; r < 8; ++r) {
        const int m    = m0 + wm + mi * 16 + g * 8 + r;
        const int bidx = m >> 10, l = m & 1023;
        Out[(((size_t)(bidx * NHEAD + h)) * SEQ + l) * DHEAD + dd] =
            (__bf16)(acc[mi][ni][r] + bias);
      }
    }
  }
}

// =====================================================================
// Kernel 2: flash attention per (b, h).  Block = 4 waves, 128 queries
// (32 per wave).  K tile staged via async global->LDS (ASYNCcnt);
// V transposed into LDS while the async copy is in flight.
// =====================================================================
__global__ __launch_bounds__(128)
void attn_kernel(const __bf16* __restrict__ qh, const __bf16* __restrict__ kh,
                 const __bf16* __restrict__ vh, const int* __restrict__ mask,
                 __bf16* __restrict__ ctx)
{
  __shared__ __align__(16) __bf16 Ks[64][72];        // [key][d]
  __shared__ __align__(16) __bf16 Vt[64][72];        // transposed: [d][key]
  __shared__ __align__(16) __bf16 Ps[4][32][72];     // per-wave P tile

  const int t = threadIdx.x, wave = t >> 5, lane = t & 31;
  const int l16 = lane & 15, g = lane >> 4;
  const int h = blockIdx.y, b = blockIdx.z;
  const int bh = b * NHEAD + h;
  const size_t base = (size_t)bh * SEQ * DHEAD;
  const int q0 = blockIdx.x * 128 + wave * 32;

  // Q fragments straight from global (rows contiguous in d).
  v16bf aq[2][2];
#pragma unroll
  for (int mi = 0; mi < 2; ++mi) {
    const __bf16* rowp = qh + base + (size_t)(q0 + mi * 16 + l16) * DHEAD;
#pragma unroll
    for (int kc = 0; kc < 2; ++kc)
      aq[mi][kc] = frag_ld(rowp + kc * 32 + g * 8, rowp + kc * 32 + 16 + g * 8);
  }

  v8f o[2][4];
#pragma unroll
  for (int mi = 0; mi < 2; ++mi)
#pragma unroll
    for (int di = 0; di < 4; ++di) o[mi][di] = {};
  float mrun[2][8], lrun[2][8];
#pragma unroll
  for (int mi = 0; mi < 2; ++mi)
#pragma unroll
    for (int r = 0; r < 8; ++r) { mrun[mi][r] = -1.0e30f; lrun[mi][r] = 0.0f; }

  const float scale = 0.125f;  // 1/sqrt(64)

  for (int kb = 0; kb < SEQ / 64; ++kb) {
    __syncthreads();   // previous iteration's LDS consumers done
    if (kb + 1 < SEQ / 64) {
      __builtin_prefetch(&kh[base + (size_t)((kb + 1) * 64) * DHEAD + t * 32], 0, 0);
      __builtin_prefetch(&vh[base + (size_t)((kb + 1) * 64) * DHEAD + t * 32], 0, 0);
    }
#if HAVE_ASYNC_LDS
    // K tile: async DMA global -> LDS, no VGPR round-trip.
#pragma unroll
    for (int i = 0; i < 4; ++i) {
      const int chunk = t + i * 128;
      const int row   = chunk >> 3;
      const int col   = (chunk & 7) * 8;
      ASYNC_B128(&kh[base + (size_t)(kb * 64 + row) * DHEAD + col], &Ks[row][col]);
    }
#endif
    // V tile: load + transpose into Vt (overlaps the async K copy).
#pragma unroll
    for (int i = 0; i < 4; ++i) {
      const int chunk = t + i * 128;        // 0..511
      const int row   = chunk >> 3;         // key 0..63
      const int col   = (chunk & 7) * 8;    // d
      H8 vv;
      vv.u = *(const u32x4*)&vh[base + (size_t)(kb * 64 + row) * DHEAD + col];
#if !HAVE_ASYNC_LDS
      H8 kv;
      kv.u = *(const u32x4*)&kh[base + (size_t)(kb * 64 + row) * DHEAD + col];
      *(u32x4*)&Ks[row][col] = kv.u;
#endif
#pragma unroll
      for (int j = 0; j < 8; ++j) Vt[col + j][row] = vv.h[j];
    }
#if HAVE_ASYNC_LDS
    WAIT_ASYNC();
#endif
    __syncthreads();

    // S = Q K^T  (32x64 per wave).
    v8f s[2][4];
#pragma unroll
    for (int mi = 0; mi < 2; ++mi)
#pragma unroll
      for (int ni = 0; ni < 4; ++ni) {
        v8f a = {};
        const __bf16* rowp = &Ks[ni * 16 + l16][0];
#pragma unroll
        for (int kc = 0; kc < 2; ++kc) {
          v16bf bk = frag_ld(rowp + kc * 32 + g * 8, rowp + kc * 32 + 16 + g * 8);
          a = wmma_bf16(aq[mi][kc], bk, a);
        }
        s[mi][ni] = a;
      }

    // scale + mask
#pragma unroll
    for (int mi = 0; mi < 2; ++mi)
#pragma unroll
      for (int ni = 0; ni < 4; ++ni) {
        const int coln = kb * 64 + ni * 16 + l16;
#pragma unroll
        for (int r = 0; r < 8; ++r) {
          const int rowq = q0 + mi * 16 + g * 8 + r;
          const float sc = s[mi][ni][r] * scale;
          const int mval = mask[((size_t)b * SEQ + rowq) * SEQ + coln];
          s[mi][ni][r] = (mval == 0) ? NEGV : sc;
        }
      }

    // Online softmax (rows span the 16 lanes of each half-wave).
#pragma unroll
    for (int mi = 0; mi < 2; ++mi)
#pragma unroll
      for (int r = 0; r < 8; ++r) {
        float smax = -1.0e30f;
#pragma unroll
        for (int ni = 0; ni < 4; ++ni) smax = fmaxf(smax, s[mi][ni][r]);
#pragma unroll
        for (int d = 1; d < 16; d <<= 1) smax = fmaxf(smax, __shfl_xor(smax, d, 32));
        const float mold = mrun[mi][r];
        const float mnew = fmaxf(mold, smax);
        const float corr = __expf(mold - mnew);
        float rsum = 0.0f;
#pragma unroll
        for (int ni = 0; ni < 4; ++ni) {
          const float p = __expf(s[mi][ni][r] - mnew);
          s[mi][ni][r] = p;
          rsum += p;
        }
#pragma unroll
        for (int d = 1; d < 16; d <<= 1) rsum += __shfl_xor(rsum, d, 32);
        mrun[mi][r] = mnew;
        lrun[mi][r] = lrun[mi][r] * corr + rsum;
#pragma unroll
        for (int di = 0; di < 4; ++di) o[mi][di][r] *= corr;
      }

    // P -> per-wave LDS (C layout -> memory) so it can re-enter as A fragments.
#pragma unroll
    for (int mi = 0; mi < 2; ++mi)
#pragma unroll
      for (int ni = 0; ni < 4; ++ni)
#pragma unroll
        for (int r = 0; r < 8; ++r)
          Ps[wave][mi * 16 + g * 8 + r][ni * 16 + l16] = (__bf16)s[mi][ni][r];
    __syncthreads();

    // O += P @ V
#pragma unroll
    for (int mi = 0; mi < 2; ++mi) {
      v16bf ap[2];
      const __bf16* prow = &Ps[wave][mi * 16 + l16][0];
#pragma unroll
      for (int kc = 0; kc < 2; ++kc)
        ap[kc] = frag_ld(prow + kc * 32 + g * 8, prow + kc * 32 + 16 + g * 8);
#pragma unroll
      for (int di = 0; di < 4; ++di) {
        const __bf16* vrow = &Vt[di * 16 + l16][0];
#pragma unroll
        for (int kc = 0; kc < 2; ++kc) {
          v16bf bv_ = frag_ld(vrow + kc * 32 + g * 8, vrow + kc * 32 + 16 + g * 8);
          o[mi][di] = wmma_bf16(ap[kc], bv_, o[mi][di]);
        }
      }
    }
  }

  // Normalize and store to concat-head layout [B, L, H*64] (bf16).
#pragma unroll
  for (int mi = 0; mi < 2; ++mi)
#pragma unroll
    for (int di = 0; di < 4; ++di)
#pragma unroll
      for (int r = 0; r < 8; ++r) {
        const int rowq = q0 + mi * 16 + g * 8 + r;
        const int col  = h * DHEAD + di * 16 + l16;
        const float val = o[mi][di][r] / lrun[mi][r];
        ctx[((size_t)(b * SEQ) + rowq) * D_MODEL + col] = (__bf16)val;
      }
}

// =====================================================================
// Kernel 3: output projection.  out[8192,1024] f32 = ctx(bf16) @ Wo^T + bo
// Same register-prefetch pipeline as kernel 1.
// =====================================================================
__global__ __launch_bounds__(128)
void out_proj_kernel(const __bf16* __restrict__ ctx, const float* __restrict__ wo,
                     const float* __restrict__ bo, float* __restrict__ out)
{
  __shared__ __align__(16) __bf16 As[128][LDSK];
  __shared__ __align__(16) __bf16 Bs[128][LDSK];

  const int m0   = blockIdx.x * 128;
  const int n0   = blockIdx.y * 128;
  const int t    = threadIdx.x;
  const int wave = t >> 5;
  const int lane = t & 31;
  const int l16  = lane & 15;
  const int g    = lane >> 4;
  const int wm   = (wave >> 1) * 64;
  const int wn   = (wave & 1) * 64;

  v8f acc[4][4];
#pragma unroll
  for (int mi = 0; mi < 4; ++mi)
#pragma unroll
    for (int ni = 0; ni < 4; ++ni) acc[mi][ni] = {};

  u32x4 pa4[4];
  f32x4 pb[8];
  auto tile_load = [&](int k0) {
#pragma unroll
    for (int i = 0; i < 4; ++i) {
      const int elem = (t + i * 128) * 8;
      const int row  = elem >> 5;
      const int col  = elem & 31;
      pa4[i] = *(const u32x4*)&ctx[(size_t)(m0 + row) * D_MODEL + k0 + col];
    }
#pragma unroll
    for (int i = 0; i < 8; ++i) {
      const int elem = (t + i * 128) * 4;
      const int row  = elem >> 5;
      const int col  = elem & 31;
      pb[i] = *(const f32x4*)&wo[(size_t)(n0 + row) * D_MODEL + k0 + col];
    }
  };
  tile_load(0);

  for (int k0 = 0; k0 < D_MODEL; k0 += 32) {
    __syncthreads();
#pragma unroll
    for (int i = 0; i < 4; ++i) {
      const int elem = (t + i * 128) * 8;
      const int row  = elem >> 5;
      const int col  = elem & 31;
      *(u32x4*)&As[row][col] = pa4[i];
    }
#pragma unroll
    for (int i = 0; i < 8; ++i) {
      const int elem = (t + i * 128) * 4;
      const int row  = elem >> 5;
      const int col  = elem & 31;
      H4 hb;
#pragma unroll
      for (int j = 0; j < 4; ++j) hb.h[j] = (__bf16)pb[i][j];
      *(u32x2*)&Bs[row][col] = hb.u;
    }
    __syncthreads();
    if (k0 + 32 < D_MODEL) tile_load(k0 + 32);  // hide behind WMMAs

    v16bf af[4], bfr[4];
#pragma unroll
    for (int mi = 0; mi < 4; ++mi) {
      const __bf16* rowp = &As[wm + mi * 16 + l16][0];
      af[mi] = frag_ld(rowp + g * 8, rowp + 16 + g * 8);
    }
#pragma unroll
    for (int ni = 0; ni < 4; ++ni) {
      const __bf16* rowp = &Bs[wn + ni * 16 + l16][0];
      bfr[ni] = frag_ld(rowp + g * 8, rowp + 16 + g * 8);
    }
#pragma unroll
    for (int mi = 0; mi < 4; ++mi)
#pragma unroll
      for (int ni = 0; ni < 4; ++ni)
        acc[mi][ni] = wmma_bf16(af[mi], bfr[ni], acc[mi][ni]);
  }

#pragma unroll
  for (int mi = 0; mi < 4; ++mi)
#pragma unroll
    for (int ni = 0; ni < 4; ++ni) {
      const int n = n0 + wn + ni * 16 + l16;
      const float bias = bo[n];
#pragma unroll
      for (int r = 0; r < 8; ++r) {
        const int m = m0 + wm + mi * 16 + g * 8 + r;
        out[(size_t)m * D_MODEL + n] = acc[mi][ni][r] + bias;
      }
    }
}

// =====================================================================
// Host-side launch
// =====================================================================
extern "C" void kernel_launch(void* const* d_in, const int* in_sizes, int n_in,
                              void* d_out, int out_size, void* d_ws, size_t ws_size,
                              hipStream_t stream) {
  (void)in_sizes; (void)n_in; (void)out_size; (void)ws_size;

  const float* q    = (const float*)d_in[0];
  const float* k    = (const float*)d_in[1];
  const float* v    = (const float*)d_in[2];
  const int*   mask = (const int*)d_in[3];
  const float* wq   = (const float*)d_in[4];
  const float* bq   = (const float*)d_in[5];
  const float* wk   = (const float*)d_in[6];
  const float* bk   = (const float*)d_in[7];
  const float* wv   = (const float*)d_in[8];
  const float* bv   = (const float*)d_in[9];
  const float* wo   = (const float*)d_in[10];
  const float* bo   = (const float*)d_in[11];
  float* out = (float*)d_out;

  // ws layout: Qh | Kh | Vh | ctx (all bf16), 16 MB each = 64 MB total.
  const size_t headElems = (size_t)BATCH * NHEAD * SEQ * DHEAD;
  __bf16* qh  = (__bf16*)d_ws;
  __bf16* khp = qh + headElems;
  __bf16* vhp = khp + headElems;
  __bf16* ctx = vhp + headElems;

  qkv_proj_kernel<<<dim3(64, 8, 3), 128, 0, stream>>>(
      q, k, v, wq, bq, wk, bk, wv, bv, qh, khp, vhp);
  attn_kernel<<<dim3(8, 16, 8), 128, 0, stream>>>(qh, khp, vhp, mask, ctx);
  out_proj_kernel<<<dim3(64, 8), 128, 0, stream>>>(ctx, wo, bo, out);
}
